// ContextAttention_28217935134953
// MI455X (gfx1250) — matrix operand order repaired and checked by reading, not compiled, not run on hardware
//
#include <hip/hip_runtime.h>
#include <hip/hip_bf16.h>
#include <math.h>

// ---------------------------------------------------------------------------
// Problem constants (from reference)
// ---------------------------------------------------------------------------
#define BB   16
#define NN   1025          // R + S*S
#define KK   257           // R + T
#define DD   768
#define HH   12
#define DHD  64
#define TT   256
#define HIDN 32
#define MQ   (BB * NN)     // 16400 rows for q / out GEMMs
#define MKV  (BB * KK)     // 4112 rows for kv GEMM

typedef __attribute__((ext_vector_type(16))) _Float16 v16h;
typedef __attribute__((ext_vector_type(8)))  _Float16 v8h;
typedef __attribute__((ext_vector_type(4)))  _Float16 v4h;
typedef __attribute__((ext_vector_type(8)))  float    v8f;

__device__ inline v16h pack16(v8h lo, v8h hi) {
    v16h r;
#pragma unroll
    for (int i = 0; i < 8; ++i) { r[i] = lo[i]; r[i + 8] = hi[i]; }
    return r;
}
__device__ inline v8f zero8() {
    v8f r;
#pragma unroll
    for (int i = 0; i < 8; ++i) r[i] = 0.0f;
    return r;
}
__device__ inline v8f wmma_f16(v16h a, v16h b, v8f c) {
    // v_wmma_f32_16x16x32_f16  (D = A*B + C, f32 accum)
    return __builtin_amdgcn_wmma_f32_16x16x32_f16(false, a, false, b,
                                                  (short)0, c, false, false);
}

// ---------------------------------------------------------------------------
// Kernel 1: learned tile pooling -> ctx (f32, [B,257,768])
// one block per (b, ctx_row). row 0 copies the R token; rows 1..256 pool a
// 2x2 patch tile with a softmax over 4 logits.
// ---------------------------------------------------------------------------
__global__ __launch_bounds__(256)
void ctx_pool_kernel(const float* __restrict__ x,
                     const float* __restrict__ logit_w,
                     float* __restrict__ ctx) {
    const int idx = blockIdx.x;
    const int b = idx / KK, row = idx % KK;
    const int tid = threadIdx.x;
    const float* xb = x + (size_t)b * NN * DD;
    float* dst = ctx + ((size_t)b * KK + row) * DD;

    if (row == 0) {
        for (int d = tid; d < DD; d += 256) dst[d] = xb[d];
        return;
    }
    const int t = row - 1, ty = t >> 4, tx = t & 15;
    const float* src[4];
#pragma unroll
    for (int dy = 0; dy < 2; ++dy)
#pragma unroll
        for (int dx = 0; dx < 2; ++dx) {
            const int tok = 1 + (2 * ty + dy) * 32 + (2 * tx + dx);
            src[dy * 2 + dx] = xb + (size_t)tok * DD;
        }

    __shared__ float sred[4][256];
    float p[4] = {0.f, 0.f, 0.f, 0.f};
    for (int d = tid; d < DD; d += 256) {
        const float lw = logit_w[d];
#pragma unroll
        for (int j = 0; j < 4; ++j) p[j] += src[j][d] * lw;
    }
#pragma unroll
    for (int j = 0; j < 4; ++j) sred[j][tid] = p[j];
    __syncthreads();
    for (int s = 128; s > 0; s >>= 1) {
        if (tid < s)
#pragma unroll
            for (int j = 0; j < 4; ++j) sred[j][tid] += sred[j][tid + s];
        __syncthreads();
    }
    __shared__ float w4[4];
    if (tid == 0) {
        float m = sred[0][0];
        for (int j = 1; j < 4; ++j) m = fmaxf(m, sred[j][0]);
        float e[4], sum = 0.f;
        for (int j = 0; j < 4; ++j) { e[j] = expf(sred[j][0] - m); sum += e[j]; }
        for (int j = 0; j < 4; ++j) w4[j] = e[j] / sum;
    }
    __syncthreads();
    const float w0 = w4[0], w1 = w4[1], w2 = w4[2], w3 = w4[3];
    for (int d = tid; d < DD; d += 256)
        dst[d] = w0 * src[0][d] + w1 * src[1][d] + w2 * src[2][d] + w3 * src[3][d];
}

// ---------------------------------------------------------------------------
// Kernel 2: WMMA GEMM  C[M,Nc] = A[M,Kd] * W[Nc,Kd]^T  (+bias)
// fp32 inputs converted to f16 in LDS; f32 accumulation via
// v_wmma_f32_16x16x32_f16. 128 threads = 4 waves; block tile 64x64,
// wave tile 32x32 (2x2 WMMA), K-step 32.
// ---------------------------------------------------------------------------
template <bool OUT_HALF>
__global__ __launch_bounds__(128)
void gemm_wmma(const float* __restrict__ A, const float* __restrict__ W,
               const float* __restrict__ bias, void* __restrict__ Cout,
               int M, int Ncols, int Kd) {
    __shared__ _Float16 Ash[64 * 40];   // 64 rows x 32 K (pad 40)
    __shared__ _Float16 Bsh[64 * 40];   // 64 cols x 32 K (pad 40)

    const int tid = threadIdx.x;
    const int wave = tid >> 5, lane = tid & 31;
    const int m0 = blockIdx.x * 64, n0 = blockIdx.y * 64;
    const int mW = (wave >> 1) * 32, nW = (wave & 1) * 32;
    const int lan15 = lane & 15;
    const bool hi16 = (lane & 16) != 0;

    v8f acc[4];
#pragma unroll
    for (int i = 0; i < 4; ++i) acc[i] = zero8();

    for (int k0 = 0; k0 < Kd; k0 += 32) {
        // cooperative load + f32->f16 convert: 512 float4 for A, 512 for B
#pragma unroll
        for (int i = 0; i < 4; ++i) {
            const int f = tid * 4 + i;
            const int row = f >> 3, c4 = (f & 7) * 4;
            float4 va = make_float4(0.f, 0.f, 0.f, 0.f);
            const int gr = m0 + row;
            if (gr < M) va = *(const float4*)(A + (size_t)gr * Kd + k0 + c4);
            v4h ha; ha[0] = (_Float16)va.x; ha[1] = (_Float16)va.y;
            ha[2] = (_Float16)va.z; ha[3] = (_Float16)va.w;
            *(v4h*)&Ash[row * 40 + c4] = ha;

            const float4 vb = *(const float4*)(W + (size_t)(n0 + row) * Kd + k0 + c4);
            v4h hb; hb[0] = (_Float16)vb.x; hb[1] = (_Float16)vb.y;
            hb[2] = (_Float16)vb.z; hb[3] = (_Float16)vb.w;
            *(v4h*)&Bsh[row * 40 + c4] = hb;
        }
        __syncthreads();

        // A fragment: lane m holds K {0..7,16..23} (low half) / {8..15,24..31}
        v16h afr[2], bfr[2];
#pragma unroll
        for (int mi = 0; mi < 2; ++mi) {
            const _Float16* ap = &Ash[(mW + mi * 16 + lan15) * 40 + (hi16 ? 8 : 0)];
            afr[mi] = pack16(*(const v8h*)ap, *(const v8h*)(ap + 16));
        }
        // B fragment: lane n holds K {0..7,16..23} (low half) / {8..15,24..31}
#pragma unroll
        for (int ni = 0; ni < 2; ++ni) {
            const _Float16* bp = &Bsh[(nW + ni * 16 + lan15) * 40 + (hi16 ? 8 : 0)];
            bfr[ni] = pack16(*(const v8h*)bp, *(const v8h*)(bp + 16));
        }
#pragma unroll
        for (int mi = 0; mi < 2; ++mi)
#pragma unroll
            for (int ni = 0; ni < 2; ++ni)
                acc[mi * 2 + ni] = wmma_f16(afr[mi], bfr[ni], acc[mi * 2 + ni]);
        __syncthreads();
    }

    // C layout: VGPR r -> M = r (+8 for lanes 16..31), N = lane%16
#pragma unroll
    for (int mi = 0; mi < 2; ++mi)
#pragma unroll
        for (int ni = 0; ni < 2; ++ni)
#pragma unroll
            for (int r = 0; r < 8; ++r) {
                const int gm = m0 + mW + mi * 16 + r + (hi16 ? 8 : 0);
                const int gn = n0 + nW + ni * 16 + lan15;
                if (gm < M) {
                    float v = acc[mi * 2 + ni][r];
                    if (OUT_HALF) {
                        ((_Float16*)Cout)[(size_t)gm * Ncols + gn] = (_Float16)v;
                    } else {
                        if (bias) v += bias[gn];
                        ((float*)Cout)[(size_t)gm * Ncols + gn] = v;
                    }
                }
            }
}

// ---------------------------------------------------------------------------
// Kernel 3: continuous-position-bias MLP: 3 -> 32 (gelu) -> 12
// one thread per (patch nP, tile kc) pair; writes biasCtx[(nP*256+kc)*12+h]
// ---------------------------------------------------------------------------
__global__ __launch_bounds__(256)
void cpb_kernel(const float* __restrict__ w1, const float* __restrict__ b1,
                const float* __restrict__ w2, const float* __restrict__ b2,
                const float* __restrict__ u_pos, const float* __restrict__ P_pos,
                const float* __restrict__ tile_centers,
                float* __restrict__ biasCtx) {
    const int idx = blockIdx.x * 256 + threadIdx.x;   // 1024*256
    const int nP = idx >> 8, kc = idx & 255;
    const float dx = P_pos[nP * 3 + 0] - (tile_centers[kc * 3 + 0] + u_pos[0]);
    const float dy = P_pos[nP * 3 + 1] - (tile_centers[kc * 3 + 1] + u_pos[1]);
    const float dz = P_pos[nP * 3 + 2] - (tile_centers[kc * 3 + 2] + u_pos[2]);
    const float f0 = (dx > 0.f ? 1.f : (dx < 0.f ? -1.f : 0.f)) * log1pf(fabsf(dx));
    const float f1 = (dy > 0.f ? 1.f : (dy < 0.f ? -1.f : 0.f)) * log1pf(fabsf(dy));
    const float f2 = -dz;
    float h[HIDN];
#pragma unroll
    for (int i = 0; i < HIDN; ++i) {
        const float a = w1[i * 3] * f0 + w1[i * 3 + 1] * f1 + w1[i * 3 + 2] * f2 + b1[i];
        h[i] = 0.5f * a * (1.0f + erff(a * 0.70710678118654752f));  // exact gelu
    }
#pragma unroll
    for (int o = 0; o < HH; ++o) {
        float a = b2[o];
#pragma unroll
        for (int i = 0; i < HIDN; ++i) a += w2[o * HIDN + i] * h[i];
        biasCtx[(size_t)idx * HH + o] = a;
    }
}

// ---------------------------------------------------------------------------
// Kernel 4: fused flash attention, one (b,h) per blockIdx.x, 8 waves.
// Per wave: 16-query tile, online softmax over 9 x 32-key chunks.
// QK^T and P*V both via v_wmma_f32_16x16x32_f16. V staged transposed in LDS;
// P round-trips a per-wave LDS buffer (s_wait_dscnt) to reach A-layout.
// ---------------------------------------------------------------------------
__global__ __launch_bounds__(256)
void attn_kernel(const _Float16* __restrict__ qbuf,
                 const _Float16* __restrict__ kvbuf,
                 const float* __restrict__ biasCtx,
                 float* __restrict__ obuf) {
    __shared__ _Float16 Vt[64 * 296];        // [dh][key], keys padded to 288
    __shared__ _Float16 Pst[8 * 16 * 40];    // per-wave P staging, 16x32 (pad 40)

    const int bh = blockIdx.x;
    const int b = bh / HH, h = bh % HH;
    const int tid = threadIdx.x, wave = tid >> 5, lane = tid & 31;
    const int lan15 = lane & 15;
    const bool hi16 = (lane & 16) != 0;
    const _Float16* kvb = kvbuf + (size_t)b * KK * (2 * DD);

    // stage V transposed: Vt[dh][key]; zero-pad keys 257..287
    for (int e = tid; e < 288 * 64; e += 256) {
        const int key = e >> 6, dh = e & 63;
        _Float16 v = (_Float16)0.0f;
        if (key < KK) v = kvb[(size_t)key * (2 * DD) + DD + h * DHD + dh];
        Vt[dh * 296 + key] = v;
    }
    __syncthreads();

    _Float16* Pw = &Pst[wave * 16 * 40];
    const float scale = 0.125f;  // DH^-0.5

    for (int qt = blockIdx.y * 8 + wave; qt < 65; qt += gridDim.y * 8) {
        const int qBase = qt * 16;
        int qrow = qBase + lan15; if (qrow > NN - 1) qrow = NN - 1;
        const _Float16* qp = qbuf + ((size_t)(b * NN + qrow)) * DD + h * DHD;
        v16h aq[2];
#pragma unroll
        for (int c = 0; c < 2; ++c) {
            const int base = c * 32 + (hi16 ? 8 : 0);
            aq[c] = pack16(*(const v8h*)(qp + base), *(const v8h*)(qp + base + 16));
        }

        float mrow[8], lrow[8];
        v8f oacc[4];
#pragma unroll
        for (int r = 0; r < 8; ++r) { mrow[r] = -INFINITY; lrow[r] = 0.f; }
#pragma unroll
        for (int n = 0; n < 4; ++n) oacc[n] = zero8();

        for (int kt = 0; kt < 9; ++kt) {
            const int k0 = kt * 32;
            v8f s[2]; s[0] = zero8(); s[1] = zero8();
#pragma unroll
            for (int c = 0; c < 2; ++c)
#pragma unroll
                for (int t2 = 0; t2 < 2; ++t2) {
                    int key = k0 + t2 * 16 + lan15; if (key > KK - 1) key = KK - 1;
                    const _Float16* kp = kvb + (size_t)key * (2 * DD) + h * DHD +
                                         c * 32 + (hi16 ? 8 : 0);
                    const v16h bk = pack16(*(const v8h*)kp, *(const v8h*)(kp + 16));
                    s[t2] = wmma_f16(aq[c], bk, s[t2]);
                }

            // scale + bias + key mask; per-row max
            float sRaw[2][8], tmax[8];
#pragma unroll
            for (int r = 0; r < 8; ++r) tmax[r] = -1e30f;
#pragma unroll
            for (int t2 = 0; t2 < 2; ++t2)
#pragma unroll
                for (int r = 0; r < 8; ++r) {
                    const int q = qBase + r + (hi16 ? 8 : 0);
                    const int key = k0 + t2 * 16 + lan15;
                    float v = s[t2][r] * scale;
                    if (q >= 1 && q < NN && key >= 1 && key < KK)
                        v += biasCtx[((size_t)(q - 1) * TT + (key - 1)) * HH + h];
                    if (key >= KK) v = -1e30f;
                    sRaw[t2][r] = v;
                    tmax[r] = fmaxf(tmax[r], v);
                }
#pragma unroll
            for (int r = 0; r < 8; ++r) {
                tmax[r] = fmaxf(tmax[r], __shfl_xor(tmax[r], 1, 32));
                tmax[r] = fmaxf(tmax[r], __shfl_xor(tmax[r], 2, 32));
                tmax[r] = fmaxf(tmax[r], __shfl_xor(tmax[r], 4, 32));
                tmax[r] = fmaxf(tmax[r], __shfl_xor(tmax[r], 8, 32));
            }
            float sf[8];
#pragma unroll
            for (int r = 0; r < 8; ++r) {
                const float nm = fmaxf(mrow[r], tmax[r]);
                sf[r] = expf(mrow[r] - nm);
                mrow[r] = nm;
            }
            float rsum[8];
#pragma unroll
            for (int r = 0; r < 8; ++r) rsum[r] = 0.f;
#pragma unroll
            for (int t2 = 0; t2 < 2; ++t2)
#pragma unroll
                for (int r = 0; r < 8; ++r) {
                    const float p = expf(sRaw[t2][r] - mrow[r]);
                    rsum[r] += p;
                    const int prow = r + (hi16 ? 8 : 0);
                    Pw[prow * 40 + t2 * 16 + lan15] = (_Float16)p;
                }
#pragma unroll
            for (int r = 0; r < 8; ++r) {
                rsum[r] += __shfl_xor(rsum[r], 1, 32);
                rsum[r] += __shfl_xor(rsum[r], 2, 32);
                rsum[r] += __shfl_xor(rsum[r], 4, 32);
                rsum[r] += __shfl_xor(rsum[r], 8, 32);
                lrow[r] = lrow[r] * sf[r] + rsum[r];
            }
#pragma unroll
            for (int n = 0; n < 4; ++n)
#pragma unroll
                for (int r = 0; r < 8; ++r) oacc[n][r] *= sf[r];

            // wait for wave-private LDS transpose of P (cross-lane via LDS)
            asm volatile("s_wait_dscnt 0" ::: "memory");

            const _Float16* pp = Pw + lan15 * 40 + (hi16 ? 8 : 0);
            const v16h pf = pack16(*(const v8h*)pp, *(const v8h*)(pp + 16));
#pragma unroll
            for (int n = 0; n < 4; ++n) {
                const _Float16* vp = &Vt[(n * 16 + lan15) * 296 + k0 + (hi16 ? 8 : 0)];
                const v16h bv = pack16(*(const v8h*)vp, *(const v8h*)(vp + 16));
                oacc[n] = wmma_f16(pf, bv, oacc[n]);
            }
        }

        // normalize + write O (f32) at [b, q, h*64 + dh]
#pragma unroll
        for (int n = 0; n < 4; ++n)
#pragma unroll
            for (int r = 0; r < 8; ++r) {
                const int q = qBase + r + (hi16 ? 8 : 0);
                if (q < NN)
                    obuf[((size_t)(b * NN + q)) * DD + h * DHD + n * 16 + lan15] =
                        oacc[n][r] / lrow[r];
            }
    }
}

// ---------------------------------------------------------------------------
// host launcher
// ---------------------------------------------------------------------------
static inline size_t align256(size_t x) { return (x + 255) & ~(size_t)255; }

extern "C" void kernel_launch(void* const* d_in, const int* in_sizes, int n_in,
                              void* d_out, int out_size, void* d_ws, size_t ws_size,
                              hipStream_t stream) {
    const float* x            = (const float*)d_in[0];
    const float* logit_w      = (const float*)d_in[1];
    const float* q_w          = (const float*)d_in[2];
    const float* kv_w         = (const float*)d_in[3];
    const float* out_w        = (const float*)d_in[4];
    const float* out_b        = (const float*)d_in[5];
    const float* cpb_w1       = (const float*)d_in[6];
    const float* cpb_b1       = (const float*)d_in[7];
    const float* cpb_w2       = (const float*)d_in[8];
    const float* cpb_b2       = (const float*)d_in[9];
    const float* u_pos        = (const float*)d_in[10];
    const float* P_pos        = (const float*)d_in[11];
    const float* tile_centers = (const float*)d_in[12];

    char* ws = (char*)d_ws;
    size_t off = 0;
    float* ctx = (float*)(ws + off);          off += align256((size_t)MKV * DD * 4);
    _Float16* qf = (_Float16*)(ws + off);     off += align256((size_t)MQ * DD * 2);
    _Float16* kvf = (_Float16*)(ws + off);    off += align256((size_t)MKV * 2 * DD * 2);
    float* biasCtx = (float*)(ws + off);      off += align256((size_t)1024 * TT * HH * 4);
    float* obuf = (float*)(ws + off);         off += align256((size_t)MQ * DD * 4);

    // 1) learned tile pooling -> ctx
    ctx_pool_kernel<<<BB * KK, 256, 0, stream>>>(x, logit_w, ctx);

    // 2) q = x @ q_w^T   (f16 out)
    gemm_wmma<true><<<dim3((MQ + 63) / 64, DD / 64), 128, 0, stream>>>(
        x, q_w, nullptr, qf, MQ, DD, DD);

    // 3) kv = ctx @ kv_w^T  (f16 out, cols [0,768)=k  [768,1536)=v)
    gemm_wmma<true><<<dim3((MKV + 63) / 64, (2 * DD) / 64), 128, 0, stream>>>(
        ctx, kv_w, nullptr, kvf, MKV, 2 * DD, DD);

    // 4) CPB bias MLP
    cpb_kernel<<<1024, 256, 0, stream>>>(cpb_w1, cpb_b1, cpb_w2, cpb_b2,
                                         u_pos, P_pos, tile_centers, biasCtx);

    // 5) fused flash attention -> obuf
    attn_kernel<<<dim3(BB * HH, 3), 256, 0, stream>>>(qf, kvf, biasCtx, obuf);

    // 6) out = obuf @ out_w^T + out_b  (f32 -> d_out)
    gemm_wmma<false><<<dim3((MQ + 63) / 64, DD / 64), 128, 0, stream>>>(
        obuf, out_w, out_b, (float*)d_out, MQ, DD, DD);
}
